// AnfisLayer_19104014532939
// MI455X (gfx1250) — matrix operand order repaired and checked by reading, not compiled, not run on hardware
//
#include <hip/hip_runtime.h>

typedef __attribute__((ext_vector_type(16))) __bf16 v16bf;
typedef __attribute__((ext_vector_type(8)))  float  v8f;

// Problem constants
#define BATCH   1024
#define NI      8
#define NMF     4
#define NRULES  65536        // 4^8
#define TILE_M  16           // batch rows per WMMA tile
#define NTILES  (BATCH / TILE_M)        // 64
#define NHC     32                      // h-chunks (h = 0..255 in 32 chunks of 8)
#define HPC     8                       // h per chunk
#define TPB     4                       // batch tiles per block (4 waves/block)

// Use CDNA5 async global->LDS copies (ASYNCcnt) for B-panel staging.
#define USE_ASYNC 1

// ws layout (bytes), all 256-aligned
#define OFF_LOA   0u                           // 64*8*32*16 bf16  = 512 KB (A-layout lo table)
#define OFF_HI    (524288u)                    // 1024*256 f32     = 1 MB
#define OFF_CWB   (1572864u)                   // 2048*32*16 bf16  = 2 MB (B-layout [cw|bias|1])
#define OFF_PART  (3670016u)                   // 2048*8*32 f32    = 2 MB partial V
#define WS_NEEDED (5767168u)

__device__ __forceinline__ unsigned short f2bf(float f) {
    unsigned u = __float_as_uint(f);
    unsigned r = u + 0x7fffu + ((u >> 16) & 1u);   // round-to-nearest-even
    return (unsigned short)(r >> 16);
}

// ---------------------------------------------------------------------------
// Kernel 1: per-batch membership products -> hi table (f32) + lo table in
// BF16 A-matrix (16x32, MxK) VGPR layout:
//   lane L<16 : M=L, halfs 0..7 = K 0..7,  halfs 8..15 = K 16..23
//   lane L+16 : M=L, halfs 0..7 = K 8..15, halfs 8..15 = K 24..31
// ---------------------------------------------------------------------------
__global__ void prep_tables(const float* __restrict__ x,
                            const float* __restrict__ mu,
                            const float* __restrict__ sigma,
                            unsigned short* __restrict__ loA,
                            float* __restrict__ hiT) {
    int b = blockIdx.x * blockDim.x + threadIdx.x;
    if (b >= BATCH) return;

    float m[NI][NMF];
#pragma unroll
    for (int i = 0; i < NI; ++i) {
        float xv = x[b * NI + i];
#pragma unroll
        for (int j = 0; j < NMF; ++j) {
            float s = fabsf(sigma[i * NMF + j]) + 1e-5f;
            float inv2v = 1.0f / (2.0f * s * s);
            float d = xv - mu[i * NMF + j];
            m[i][j] = __expf(-d * d * inv2v);
        }
    }
    float p01[16], p23[16], p45[16], p67[16];
#pragma unroll
    for (int a = 0; a < 4; ++a)
#pragma unroll
        for (int c = 0; c < 4; ++c) {
            p01[a * 4 + c] = m[0][a] * m[1][c];
            p23[a * 4 + c] = m[2][a] * m[3][c];
            p45[a * 4 + c] = m[4][a] * m[5][c];
            p67[a * 4 + c] = m[6][a] * m[7][c];
        }
    // hi[h], h = base-4 digits of inputs 0..3 (r>>8)
#pragma unroll
    for (int h = 0; h < 256; ++h)
        hiT[b * 256 + h] = p01[h >> 4] * p23[h & 15];

    // lo[k] -> BF16 A-fragment layout (k = r & 255, inputs 4..7)
    int t = b >> 4, mrow = b & 15;
#pragma unroll
    for (int k = 0; k < 256; ++k) {
        float lo = p45[k >> 4] * p67[k & 15];
        int ks   = k >> 5;
        int lane = mrow + (((k >> 3) & 1) << 4);
        int half = (k & 7) + (((k >> 4) & 1) << 3);
        loA[(((t * 8 + ks) * 32) + lane) * 16 + half] = f2bf(lo);
    }
}

// ---------------------------------------------------------------------------
// Kernel 2: [cw | bias | 1 | 0...] -> BF16 B-matrix (32x16, KxN) layout:
//   lane L<16 : N=L, halfs 0..7 = K 0..7,  halfs 8..15 = K 16..23
//   lane L+16 : N=L, halfs 0..7 = K 8..15, halfs 8..15 = K 24..31
// ---------------------------------------------------------------------------
__global__ void prep_cwb(const float* __restrict__ cw,
                         const float* __restrict__ bias,
                         unsigned short* __restrict__ cwb) {
    int id = blockIdx.x * blockDim.x + threadIdx.x;   // 0 .. 65535
    if (id >= (NRULES / 32) * 32) return;
    int kb = id >> 5, lane = id & 31;
    int n  = lane & 15;
    int kh = lane >> 4;
    unsigned short* dst = cwb + (size_t)(kb * 32 + lane) * 16;
#pragma unroll
    for (int hh = 0; hh < 16; ++hh) {
        int k = (hh & 7) + 8 * kh + ((hh >> 3) << 4);
        int r = kb * 32 + k;
        float v;
        if (n < 8)       v = cw[(size_t)r * NI + n];
        else if (n == 8) v = bias[r];
        else if (n == 9) v = 1.0f;   // ones column -> V[9] = sum_r w  (denominator)
        else             v = 0.0f;
        dst[hh] = f2bf(v);
    }
}

// ---------------------------------------------------------------------------
// Kernel 3: main WMMA contraction.
//   Block = 4 waves = 4 batch tiles sharing one h-chunk. The 8 KB B panel per
//   h is staged once per block into LDS (double-buffered) with CDNA5 async
//   global->LDS copies; waves consume it via ds_load_b128 with a register
//   double-buffer so each v_wmma overlaps the next fragment's LDS load.
// ---------------------------------------------------------------------------
__global__ void __launch_bounds__(32 * TPB)
anfis_wmma(const unsigned short* __restrict__ loA,
           const float* __restrict__ hiT,
           const unsigned short* __restrict__ cwb,
           float* __restrict__ partial) {
    __shared__ __align__(32) unsigned short Bbuf[2][8 * 32 * 16];   // 2 x 8 KB

    int tb   = blockIdx.x >> 5;       // tile group 0..15
    int hc   = blockIdx.x & 31;       // h-chunk   0..31
    int wave = threadIdx.x >> 5;      // 0..3
    int lane = threadIdx.x & 31;
    int t    = tb * TPB + wave;       // batch tile 0..63
    int tid  = threadIdx.x;           // 0..127

    // Load the 8 A fragments (16x256 lo panel for this tile) once.
    const v16bf* ap = (const v16bf*)loA + (size_t)t * 8 * 32 + lane;
    v16bf A[8];
#pragma unroll
    for (int ks = 0; ks < 8; ++ks) A[ks] = ap[ks * 32];

    int halfsel = lane >> 4;          // C/D layout: VGPR j holds row j + 8*halfsel
    const float* hp = hiT + (size_t)(t * TILE_M + 8 * halfsel) * 256 + hc * HPC;

    // B panels for this h-chunk: 8 panels x 8192 bytes, contiguous in cwb.
    const char* gb0 = (const char*)cwb + (size_t)(hc * HPC) * 8192;

    // Group-segment-relative LDS byte addresses (generic ptr low 32 bits).
    unsigned ldsA[2];
    ldsA[0] = (unsigned)(uintptr_t)&Bbuf[0][0];
    ldsA[1] = (unsigned)(uintptr_t)&Bbuf[1][0];

#if USE_ASYNC
    // Stage one 8 KB panel: 512 x 16B chunks, 128 threads x 4 chunks each.
    auto stage = [&](int buf, int hpanel) {
#pragma unroll
        for (int i = 0; i < 4; ++i) {
            unsigned off   = (unsigned)(tid + i * 128) * 16u;
            unsigned laddr = ldsA[buf] + off;
            const char* gb = gb0 + (size_t)hpanel * 8192;
            asm volatile("global_load_async_to_lds_b128 %0, %1, %2"
                         :
                         : "v"(laddr), "v"(off), "s"(gb)
                         : "memory");
        }
    };
    stage(0, 0);
    asm volatile("s_wait_asynccnt 0x0" ::: "memory");
    __syncthreads();
#else
    int4 st[4];
    auto stage_issue = [&](int hpanel, int4* r) {
        const int4* g = (const int4*)(gb0 + (size_t)hpanel * 8192);
#pragma unroll
        for (int i = 0; i < 4; ++i) r[i] = g[tid + i * 128];
    };
    auto stage_commit = [&](int buf, const int4* r) {
        int4* l = (int4*)&Bbuf[buf][0];
#pragma unroll
        for (int i = 0; i < 4; ++i) l[tid + i * 128] = r[i];
    };
    stage_issue(0, st);
    stage_commit(0, st);
    __syncthreads();
#endif

    v8f accV = {};
    for (int hh = 0; hh < HPC; ++hh) {
        int cur = hh & 1;
#if USE_ASYNC
        // Async writes into Bbuf[cur^1]: WAR-safe, its last readers finished
        // before the barrier that ended the previous iteration.
        if (hh + 1 < HPC) stage(cur ^ 1, hh + 1);
#else
        int4 nx[4];
        if (hh + 1 < HPC) stage_issue(hh + 1, nx);
#endif
        // Register-double-buffered B fragments: load ks+1 before consuming ks
        // so each v_wmma overlaps the next ds_load pair.
        v8f acc = {};
        v16bf Bc = *(const v16bf*)&Bbuf[cur][lane * 16];
#pragma unroll
        for (int ks = 0; ks < 8; ++ks) {
            v16bf Bn;
            if (ks + 1 < 8)
                Bn = *(const v16bf*)&Bbuf[cur][((ks + 1) * 32 + lane) * 16];
            acc = __builtin_amdgcn_wmma_f32_16x16x32_bf16(
                      false, A[ks], false, Bc, (short)0, acc, false, false);
            Bc = Bn;
        }
#pragma unroll
        for (int j = 0; j < 8; ++j) {
            float hv = hp[j * 256 + hh];         // hi[t*16 + j + 8*halfsel][h]
            accV[j] += hv * acc[j];
        }
#if USE_ASYNC
        if (hh + 1 < HPC) {
            asm volatile("s_wait_asynccnt 0x0" ::: "memory");
            __syncthreads();
        }
#else
        if (hh + 1 < HPC) {
            __syncthreads();                 // WAR: everyone done reading cur^1
            stage_commit(cur ^ 1, nx);
            __syncthreads();                 // RAW: stores visible to all waves
        }
#endif
    }
    // partial layout: [t*NHC+hc][j][lane] (value = V_part[b=t*16+j+8*halfsel][n=lane&15])
    float* pp = partial + ((size_t)t * NHC + hc) * 8 * 32;
#pragma unroll
    for (int j = 0; j < 8; ++j) pp[j * 32 + lane] = accV[j];
}

// ---------------------------------------------------------------------------
// Kernel 4: deterministic reduction over h-chunks + output formula.
//   out[b] = (x[b]·V[0:8] + V[8]) / (V[9] + 1e-6)
// ---------------------------------------------------------------------------
__global__ void finalize(const float* __restrict__ x,
                         const float* __restrict__ partial,
                         float* __restrict__ out) {
    int b = blockIdx.x * blockDim.x + threadIdx.x;
    if (b >= BATCH) return;
    int t = b >> 4, mrow = b & 15;
    int j = mrow & 7, halfsel = mrow >> 3;

    float V[16];
#pragma unroll
    for (int n = 0; n < 16; ++n) {
        int lane = n + 16 * halfsel;
        float s = 0.0f;
        for (int hc = 0; hc < NHC; ++hc)
            s += partial[((size_t)(t * NHC + hc) * 8 + j) * 32 + lane];
        V[n] = s;
    }
    float num = V[8];
#pragma unroll
    for (int i = 0; i < NI; ++i) num += x[b * NI + i] * V[i];
    out[b] = num / (V[9] + 1e-6f);
}

extern "C" void kernel_launch(void* const* d_in, const int* in_sizes, int n_in,
                              void* d_out, int out_size, void* d_ws, size_t ws_size,
                              hipStream_t stream) {
    const float* x     = (const float*)d_in[0];   // (1024, 8)
    const float* mu    = (const float*)d_in[1];   // (8, 4)
    const float* sigma = (const float*)d_in[2];   // (8, 4)
    const float* cw    = (const float*)d_in[3];   // (65536, 8)
    const float* bias  = (const float*)d_in[4];   // (65536,)
    float* out = (float*)d_out;                   // (1024, 1)

    char* ws = (char*)d_ws;
    unsigned short* loA  = (unsigned short*)(ws + OFF_LOA);
    float*          hiT  = (float*)         (ws + OFF_HI);
    unsigned short* cwb  = (unsigned short*)(ws + OFF_CWB);
    float*          part = (float*)         (ws + OFF_PART);

    prep_tables<<<dim3(BATCH / 256), dim3(256), 0, stream>>>(x, mu, sigma, loA, hiT);
    prep_cwb<<<dim3(NRULES / 256), dim3(256), 0, stream>>>(cw, bias, cwb);
    anfis_wmma<<<dim3((NTILES / TPB) * NHC), dim3(32 * TPB), 0, stream>>>(loA, hiT, cwb, part);
    finalize<<<dim3(BATCH / 256), dim3(256), 0, stream>>>(x, part, out);
}
